// MoELayer_32469952757764
// MI455X (gfx1250) — compile-verified, matrix operand order checked by reading, not run on hardware
//
#include <hip/hip_runtime.h>
#include <hip/hip_bf16.h>
#include <cstddef>

// ---- problem constants (from reference) ----
#define T_TOK   16384      // B*S
#define D_MODEL 1024
#define D_FF    4096
#define N_EXP   8
#define CAPACITY 3277      // ceil(16384*1.6/8)
#define CAP_PAD  3328      // 26 * 128, padded with sentinel T_TOK

// ---- GEMM tiling ----
#define BM 128
#define BN 128
#define BK 32
#define LDA 40             // BK + 8 pad (bf16 elems) -> 80B row stride, 16B aligned
#define LDB 40

typedef __bf16 bf16x4  __attribute__((ext_vector_type(4)));
typedef __bf16 bf16x8  __attribute__((ext_vector_type(8)));
typedef __bf16 bf16x16 __attribute__((ext_vector_type(16)));
typedef float  floatx8 __attribute__((ext_vector_type(8)));

static __device__ __forceinline__ __bf16 f2bf(float f) {
  union { float f; unsigned u; } in; in.f = f;
  unsigned r = (in.u + 0x7FFFu + ((in.u >> 16) & 1u)) >> 16;  // round-nearest-even
  union { unsigned short u; __bf16 b; } out; out.u = (unsigned short)r;
  return out.b;
}

static __device__ __forceinline__ float gelu_exact(float v) {
  return 0.5f * v * (1.0f + erff(v * 0.70710678118654752f));
}

// Generic LDS pointer -> wave-relative 32-bit LDS byte offset (flat LDS aperture
// truncates to addr[31:0], ISA 10.2).
static __device__ __forceinline__ unsigned lds_addr32(const void* p) {
  return (unsigned)(unsigned long long)p;
}

// gfx1250 async copy: 32 contiguous bytes global -> LDS per lane, two
// GLOBAL_LOAD_ASYNC_TO_LDS_B128 (GVS mode: SGPR base + 32-bit VGPR offset).
// INST_OFFSET is added on BOTH the LDS and global side (ISA 08, 4.4).
static __device__ __forceinline__ void async_copy32(unsigned lds_off, unsigned gbyte_off,
                                                    const void* sbase) {
  asm volatile("global_load_async_to_lds_b128 %0, %1, %2\n\t"
               "global_load_async_to_lds_b128 %0, %1, %2 offset:16"
               :: "v"(lds_off), "v"(gbyte_off), "s"(sbase) : "memory");
}
static __device__ __forceinline__ void async_wait4() {
  asm volatile("s_wait_asynccnt 0x4" ::: "memory");
}
static __device__ __forceinline__ void async_wait0() {
  asm volatile("s_wait_asynccnt 0x0" ::: "memory");
}

union FragU { bf16x16 v; bf16x8 h[2]; };

// ISA 16-bit A/B fragment: lane&15 selects row(A)/col(B), lane>>4 selects K-half.
// Per lane: 8 bf16 at [outer][khalf*8] and 8 bf16 at [outer][16+khalf*8].
static __device__ __forceinline__ bf16x16 load_frag_lds(const __bf16* p) {
  FragU r;
  r.h[0] = *reinterpret_cast<const bf16x8*>(p);        // K-run 0
  r.h[1] = *reinterpret_cast<const bf16x8*>(p + 16);   // K-run 1 (+16 elems)
  return r.v;
}

// ---------------------------------------------------------------------------
// 0) zero d_out (dropped tokens must read 0; harness poisons buffers)
__global__ __launch_bounds__(256) void moe_zero_kernel(float4* __restrict__ out, int n4) {
  int i = blockIdx.x * 256 + threadIdx.x;
  if (i < n4) out[i] = make_float4(0.f, 0.f, 0.f, 0.f);
}

// ---------------------------------------------------------------------------
// 1) x (f32) -> xbf (bf16), with an extra all-zero row T_TOK = the reference's
//    x_pad row, so the capacity-padded gather is branchless.
__global__ __launch_bounds__(256)
void moe_convert_x_kernel(const float* __restrict__ x, __bf16* __restrict__ xbf) {
  const size_t i4 = ((size_t)blockIdx.x * 256 + threadIdx.x) * 4;
  if (i4 >= (size_t)(T_TOK + 1) * D_MODEL) return;
  bf16x4 o;
  if (i4 < (size_t)T_TOK * D_MODEL) {
    const float4 v = *reinterpret_cast<const float4*>(x + i4);
    o[0] = f2bf(v.x); o[1] = f2bf(v.y); o[2] = f2bf(v.z); o[3] = f2bf(v.w);
  } else {
    o[0] = o[1] = o[2] = o[3] = f2bf(0.f);
  }
  *reinterpret_cast<bf16x4*>(xbf + i4) = o;
}

// ---------------------------------------------------------------------------
// 2) W [K][N] f32 -> Wt [N][K] bf16 (32x32 LDS tile; makes GEMM B-operand a
//    straight row-major-in-K bf16 copy, killing the scalar ds_store_b16 path)
__global__ __launch_bounds__(256)
void moe_transpose_bf16_kernel(const float* __restrict__ W, __bf16* __restrict__ Wt,
                               int K, int N) {
  __shared__ __bf16 tile[32][40];     // 80B row stride: 8B-aligned b64 reads
  const int t = threadIdx.x;
  const int k0 = blockIdx.y * 32, n0 = blockIdx.x * 32;
  {
    const int kr = t >> 3, n4 = (t & 7) << 2;   // coalesced f32 reads along n
    const float4 v = *reinterpret_cast<const float4*>(W + (size_t)(k0 + kr) * N + n0 + n4);
    tile[n4 + 0][kr] = f2bf(v.x);
    tile[n4 + 1][kr] = f2bf(v.y);
    tile[n4 + 2][kr] = f2bf(v.z);
    tile[n4 + 3][kr] = f2bf(v.w);
  }
  __syncthreads();
  {
    const int nr = t >> 3, k4 = (t & 7) << 2;   // coalesced bf16x4 writes along k
    bf16x4 o;
    o[0] = tile[nr][k4 + 0]; o[1] = tile[nr][k4 + 1];
    o[2] = tile[nr][k4 + 2]; o[3] = tile[nr][k4 + 3];
    *reinterpret_cast<bf16x4*>(Wt + (size_t)(n0 + nr) * K + k0 + k4) = o;
  }
}

// ---------------------------------------------------------------------------
// 3) router: logits = x @ rw^T + rb + noise*0.02; top-2 -> expert bit mask.
__global__ __launch_bounds__(256)
void moe_router_kernel(const float* __restrict__ x, const float* __restrict__ rw,
                       const float* __restrict__ rb, const float* __restrict__ noise,
                       int* __restrict__ masks) {
  __shared__ float rws[N_EXP * D_MODEL];
  const int tid = threadIdx.x;
  for (int i = tid; i < N_EXP * D_MODEL; i += 256) rws[i] = rw[i];
  __syncthreads();

  const int lane = tid & 31, wave = tid >> 5;
  const int t = blockIdx.x * 8 + wave;
  const float* xr = x + (size_t)t * D_MODEL;

  float acc[N_EXP] = {};
  for (int d = lane; d < D_MODEL; d += 32) {
    const float xv = xr[d];
#pragma unroll
    for (int e = 0; e < N_EXP; ++e) acc[e] = fmaf(xv, rws[e * D_MODEL + d], acc[e]);
  }
#pragma unroll
  for (int e = 0; e < N_EXP; ++e)
#pragma unroll
    for (int off = 16; off; off >>= 1) acc[e] += __shfl_xor(acc[e], off, 32);

  if (lane == 0) {
    float lg[N_EXP];
#pragma unroll
    for (int e = 0; e < N_EXP; ++e)
      lg[e] = acc[e] + rb[e] + noise[(size_t)t * N_EXP + e] * 0.02f;
    int m1 = 0;
    for (int e = 1; e < N_EXP; ++e) if (lg[e] > lg[m1]) m1 = e;   // ties: lowest idx
    int m2 = -1;
    for (int e = 0; e < N_EXP; ++e) {
      if (e == m1) continue;
      if (m2 < 0 || lg[e] > lg[m2]) m2 = e;
    }
    masks[t] = (1 << m1) | (1 << m2);
  }
}

// ---------------------------------------------------------------------------
// 4) ordered compaction: first CAPACITY tokens in token order per expert,
//    sentinel T_TOK elsewhere (points at the zero row of xbf).
__global__ __launch_bounds__(32)
void moe_compact_kernel(const int* __restrict__ masks, int* __restrict__ idxbuf) {
  const int e = blockIdx.x;
  const int lane = threadIdx.x;
  int* idx = idxbuf + e * CAP_PAD;
  int cnt = 0;
  for (int base = 0; base < T_TOK; base += 32) {
    const int p = (masks[base + lane] >> e) & 1;
    const unsigned bal = (unsigned)__ballot(p);   // wave32: low 32 bits
    const int pos = cnt + __popc(bal & ((1u << lane) - 1u));
    if (p && pos < CAPACITY) idx[pos] = base + lane;
    cnt += __popc(bal);
  }
  const int start = cnt < CAPACITY ? cnt : CAPACITY;
  for (int i = start + lane; i < CAP_PAD; i += 32) idx[i] = T_TOK;
}

// ---------------------------------------------------------------------------
// 5) GEMM1: H[cap,DFF] = bf16( gelu( gather(xbf) @ W1t^T + b1 ) )
//    128x128x32 tiles, 8 waves (wave = 32x64 = 2x4 WMMA), async double-buffered
//    LDS staging via GLOBAL_LOAD_ASYNC_TO_LDS_B128.
__global__ __launch_bounds__(256)
void moe_gemm1_kernel(const __bf16* __restrict__ xbf, const int* __restrict__ idx,
                      const __bf16* __restrict__ W1t, const float* __restrict__ b1e,
                      __bf16* __restrict__ H) {
  __shared__ __bf16 As[2][BM * LDA];
  __shared__ __bf16 Bs[2][BN * LDB];

  const int tid = threadIdx.x;
  const int lane = tid & 31, wave = tid >> 5;
  const int wm = wave & 3, wn = wave >> 2;          // 4x2 wave grid
  const int l15 = lane & 15, khalf = lane >> 4;
  const int m0 = blockIdx.y * BM, n0 = blockIdx.x * BN;

  floatx8 acc[2][4] = {};

  // staging: 2 threads/row, each copies 32B (two async b128)
  const int arow = tid >> 1, ahalf = tid & 1;
  const int token = idx[m0 + arow];                 // 0..T_TOK (T_TOK = zero row)
  unsigned a_g = ((unsigned)token * D_MODEL + ahalf * 16) * 2;          // bytes
  unsigned b_g = ((unsigned)(n0 + arow) * D_MODEL + ahalf * 16) * 2;    // bytes
  const unsigned a_l[2] = { lds_addr32(&As[0][arow * LDA + ahalf * 16]),
                            lds_addr32(&As[1][arow * LDA + ahalf * 16]) };
  const unsigned b_l[2] = { lds_addr32(&Bs[0][arow * LDB + ahalf * 16]),
                            lds_addr32(&Bs[1][arow * LDB + ahalf * 16]) };

  async_copy32(a_l[0], a_g, xbf);                    // prologue: tile 0
  async_copy32(b_l[0], b_g, W1t);

  int buf = 0;
  for (int k0 = 0; k0 < D_MODEL; k0 += BK) {
    if (k0 + BK < D_MODEL) {                         // prefetch tile k0+BK
      a_g += BK * 2; b_g += BK * 2;
      async_copy32(a_l[buf ^ 1], a_g, xbf);
      async_copy32(b_l[buf ^ 1], b_g, W1t);
      async_wait4();                                 // 4 newest may remain
    } else {
      async_wait0();
    }
    __syncthreads();

    bf16x16 af[2], bfr[4];
#pragma unroll
    for (int mi = 0; mi < 2; ++mi)
      af[mi] = load_frag_lds(&As[buf][(wm * 32 + mi * 16 + l15) * LDA + khalf * 8]);
#pragma unroll
    for (int ni = 0; ni < 4; ++ni)
      bfr[ni] = load_frag_lds(&Bs[buf][(wn * 64 + ni * 16 + l15) * LDB + khalf * 8]);
#pragma unroll
    for (int mi = 0; mi < 2; ++mi)
#pragma unroll
      for (int ni = 0; ni < 4; ++ni)
        acc[mi][ni] = __builtin_amdgcn_wmma_f32_16x16x32_bf16(
            false, af[mi], false, bfr[ni], (short)0, acc[mi][ni], false, false);
    __syncthreads();
    buf ^= 1;
  }

  // epilogue: +b1, exact GELU, bf16 store to H
  const int mofs = khalf * 8;   // C layout: lanes 16-31 hold M=8..15
#pragma unroll
  for (int mi = 0; mi < 2; ++mi) {
    const int rowb = m0 + wm * 32 + mi * 16 + mofs;
#pragma unroll
    for (int ni = 0; ni < 4; ++ni) {
      const int col = n0 + wn * 64 + ni * 16 + l15;
      const float bias = b1e[col];
#pragma unroll
      for (int r = 0; r < 8; ++r) {
        const float v = acc[mi][ni][r] + bias;
        H[(size_t)(rowb + r) * D_FF + col] = f2bf(gelu_exact(v));
      }
    }
  }
}

// ---------------------------------------------------------------------------
// 6) GEMM2: out[token] = H @ W2t^T + b2 (scatter; sequential expert launches
//    give the reference's overwrite order; sentinel rows skipped)
__global__ __launch_bounds__(256)
void moe_gemm2_kernel(const __bf16* __restrict__ H, const __bf16* __restrict__ W2t,
                      const float* __restrict__ b2e, const int* __restrict__ idx,
                      float* __restrict__ out) {
  __shared__ __bf16 As[2][BM * LDA];
  __shared__ __bf16 Bs[2][BN * LDB];

  const int tid = threadIdx.x;
  const int lane = tid & 31, wave = tid >> 5;
  const int wm = wave & 3, wn = wave >> 2;
  const int l15 = lane & 15, khalf = lane >> 4;
  const int m0 = blockIdx.y * BM, n0 = blockIdx.x * BN;

  floatx8 acc[2][4] = {};

  const int arow = tid >> 1, ahalf = tid & 1;
  unsigned a_g = ((unsigned)(m0 + arow) * D_FF + ahalf * 16) * 2;
  unsigned b_g = ((unsigned)(n0 + arow) * D_FF + ahalf * 16) * 2;
  const unsigned a_l[2] = { lds_addr32(&As[0][arow * LDA + ahalf * 16]),
                            lds_addr32(&As[1][arow * LDA + ahalf * 16]) };
  const unsigned b_l[2] = { lds_addr32(&Bs[0][arow * LDB + ahalf * 16]),
                            lds_addr32(&Bs[1][arow * LDB + ahalf * 16]) };

  async_copy32(a_l[0], a_g, H);
  async_copy32(b_l[0], b_g, W2t);

  int buf = 0;
  for (int k0 = 0; k0 < D_FF; k0 += BK) {
    if (k0 + BK < D_FF) {
      a_g += BK * 2; b_g += BK * 2;
      async_copy32(a_l[buf ^ 1], a_g, H);
      async_copy32(b_l[buf ^ 1], b_g, W2t);
      async_wait4();
    } else {
      async_wait0();
    }
    __syncthreads();

    bf16x16 af[2], bfr[4];
#pragma unroll
    for (int mi = 0; mi < 2; ++mi)
      af[mi] = load_frag_lds(&As[buf][(wm * 32 + mi * 16 + l15) * LDA + khalf * 8]);
#pragma unroll
    for (int ni = 0; ni < 4; ++ni)
      bfr[ni] = load_frag_lds(&Bs[buf][(wn * 64 + ni * 16 + l15) * LDB + khalf * 8]);
#pragma unroll
    for (int mi = 0; mi < 2; ++mi)
#pragma unroll
      for (int ni = 0; ni < 4; ++ni)
        acc[mi][ni] = __builtin_amdgcn_wmma_f32_16x16x32_bf16(
            false, af[mi], false, bfr[ni], (short)0, acc[mi][ni], false, false);
    __syncthreads();
    buf ^= 1;
  }

  // epilogue: +b2, scatter to out by token id
  const int mofs = khalf * 8;
#pragma unroll
  for (int mi = 0; mi < 2; ++mi) {
    const int rowb = m0 + wm * 32 + mi * 16 + mofs;
    int tok[8];
#pragma unroll
    for (int r = 0; r < 8; ++r) tok[r] = idx[rowb + r];
#pragma unroll
    for (int ni = 0; ni < 4; ++ni) {
      const int col = n0 + wn * 64 + ni * 16 + l15;
      const float bias = b2e[col];
#pragma unroll
      for (int r = 0; r < 8; ++r)
        if (tok[r] < T_TOK)
          out[(size_t)tok[r] * D_MODEL + col] = acc[mi][ni][r] + bias;
    }
  }
}

// ---------------------------------------------------------------------------
extern "C" void kernel_launch(void* const* d_in, const int* in_sizes, int n_in,
                              void* d_out, int out_size, void* d_ws, size_t ws_size,
                              hipStream_t stream) {
  (void)in_sizes; (void)n_in; (void)out_size; (void)ws_size;
  const float* x     = (const float*)d_in[0];  // [T, D]
  const float* noise = (const float*)d_in[1];  // [T, E]
  const float* rw    = (const float*)d_in[2];  // [E, D]
  const float* rb    = (const float*)d_in[3];  // [E]
  const float* w1    = (const float*)d_in[4];  // [E, D, DFF]
  const float* b1    = (const float*)d_in[5];  // [E, DFF]
  const float* w2    = (const float*)d_in[6];  // [E, DFF, D]
  const float* b2    = (const float*)d_in[7];  // [E, D]
  float* out = (float*)d_out;

  // workspace layout (64KB-aligned regions):
  char* ws = (char*)d_ws;
  int*    masks  = (int*)(ws + 0);              //  64 KB
  int*    idxbuf = (int*)(ws + 65536);          // 128 KB  (8*3328*4)
  __bf16* xbf    = (__bf16*)(ws + 196608);      // (T+1)*D*2 = 33.6 MB (row T = zeros)
  __bf16* Hbf    = (__bf16*)(ws + 33816576);    // 3328*4096*2 = 27.3 MB
  __bf16* W1t    = (__bf16*)(ws + 61145088);    // [DFF][D]  bf16 = 8.4 MB
  __bf16* W2t    = (__bf16*)(ws + 69599232);    // [D][DFF]  bf16 = 8.4 MB
                                                // total ~78 MB

  moe_zero_kernel<<<(T_TOK * D_MODEL / 4 + 255) / 256, 256, 0, stream>>>(
      (float4*)out, T_TOK * D_MODEL / 4);
  moe_convert_x_kernel<<<((T_TOK + 1) * D_MODEL / 4 + 255) / 256, 256, 0, stream>>>(x, xbf);
  moe_router_kernel<<<T_TOK / 8, 256, 0, stream>>>(x, rw, rb, noise, masks);
  moe_compact_kernel<<<N_EXP, 32, 0, stream>>>(masks, idxbuf);

  // Sequential expert launches on `stream`: later experts overwrite earlier
  // ones for tokens routed to both -> matches reference's out.at[idx].set order.
  for (int e = 0; e < N_EXP; ++e) {
    moe_transpose_bf16_kernel<<<dim3(D_FF / 32, D_MODEL / 32), 256, 0, stream>>>(
        w1 + (size_t)e * D_MODEL * D_FF, W1t, D_MODEL, D_FF);
    moe_transpose_bf16_kernel<<<dim3(D_MODEL / 32, D_FF / 32), 256, 0, stream>>>(
        w2 + (size_t)e * D_FF * D_MODEL, W2t, D_FF, D_MODEL);
    moe_gemm1_kernel<<<dim3(D_FF / BN, CAP_PAD / BM), 256, 0, stream>>>(
        xbf, idxbuf + e * CAP_PAD, W1t, b1 + (size_t)e * D_FF, Hbf);
    moe_gemm2_kernel<<<dim3(D_MODEL / BN, CAP_PAD / BM), 256, 0, stream>>>(
        Hbf, W2t, b2 + (size_t)e * D_MODEL, idxbuf + e * CAP_PAD, out);
  }
}
